// UUIIGCMCModel_12249246728546
// MI455X (gfx1250) — matrix operand order repaired and checked by reading, not compiled, not run on hardware
//
#include <hip/hip_runtime.h>
#include <hip/hip_bf16.h>
#include <math.h>

typedef __attribute__((ext_vector_type(16))) _Float16 v16h;
typedef __attribute__((ext_vector_type(8)))  _Float16 v8h;
typedef __attribute__((ext_vector_type(8)))  float    v8f;

#define TPB 256

// ---------------- utility kernels ----------------

__global__ void k_zero(float* __restrict__ p, long long n) {
  long long i = (long long)blockIdx.x * blockDim.x + threadIdx.x;
  if (i < n) p[i] = 0.0f;
}

__global__ void k_degree(const int* __restrict__ col, long long E,
                         float* __restrict__ deg) {
  long long i = (long long)blockIdx.x * blockDim.x + threadIdx.x;
  if (i < E) atomicAdd(&deg[col[i]], 1.0f);
}

__global__ void k_rsqrt_inplace(float* __restrict__ d, long long n) {
  long long i = (long long)blockIdx.x * blockDim.x + threadIdx.x;
  if (i < n) {
    float v = d[i];
    d[i] = (v > 0.0f) ? rsqrtf(fmaxf(v, 1e-12f)) : 0.0f;
  }
}

// scatter-add propagation: y[col] += dinv[row]*dinv[col] * x[row]; 64 threads/edge
__global__ void k_propagate(const float* __restrict__ x,
                            const int* __restrict__ row,
                            const int* __restrict__ col,
                            const float* __restrict__ dinv,
                            float* __restrict__ y, long long E) {
  long long t = (long long)blockIdx.x * blockDim.x + threadIdx.x;
  long long e = t >> 6;
  int d = (int)(t & 63);
  if (e >= E) return;
  int r = row[e];
  int c = col[e];
  float nrm = dinv[r] * dinv[c];
  atomicAdd(&y[(long long)c * 64 + d], nrm * x[(long long)r * 64 + d]);
}

// convert a batch of 64x64 f32 matrices to f16, TRANSPOSED per matrix:
// dst[m][n*64+k] = src[m][k*64+n]   (B-fragment-friendly: contiguous K per column)
__global__ void k_w_to_f16_t(const float* __restrict__ s,
                             _Float16* __restrict__ d, int nmats) {
  long long t = (long long)blockIdx.x * blockDim.x + threadIdx.x;
  if (t >= (long long)nmats * 4096) return;
  int m = (int)(t >> 12);
  int idx = (int)(t & 4095);
  int n = idx >> 6;
  int k = idx & 63;
  d[t] = (_Float16)s[(long long)m * 4096 + k * 64 + n];
}

__global__ void k_relu_acc(float* __restrict__ acc,
                           const float* __restrict__ x, long long n) {
  long long i = (long long)blockIdx.x * blockDim.x + threadIdx.x;
  if (i < n) acc[i] += fmaxf(x[i], 0.0f);
}

// ---------------- WMMA GEMM: Y[M x 64] = X[M x 64] * W[64 x 64] ----------------
// Wt is the f16 weight TRANSPOSED: Wt[n*64 + k] = W[k][n].
// blockDim = 128 (4 waves). Wave j computes the 16x16 tile at columns [16j,16j+16)
// of the 16-row strip at blockIdx.x*16. K=64 -> two v_wmma_f32_16x16x32_f16.
// A frag (16x32): lanes 0-15 M=lane, K={kk+0..7, kk+16..23}; lanes 16-31 K+=8.
// B frag (32x16): lane n=lane&15, K={kk+0..7, kk+16..23}; lanes 16-31 K+=8.
__global__ void k_gemm64(const float* __restrict__ X,
                         const _Float16* __restrict__ Wt,
                         float* __restrict__ Y, int nrows) {
  int wave = threadIdx.x >> 5;
  int lane = threadIdx.x & 31;
  int m0 = blockIdx.x * 16;
  if (m0 >= nrows) return;
  int j = wave;                    // column tile 0..3
  int nloc = lane & 15;
  int hi = lane >> 4;              // 0 | 1
  int abase = hi * 8;              // A K-subgroup offset
  int kb2 = hi * 8;                // B K-subgroup offset
  const float* __restrict__ Xr = X + (long long)(m0 + nloc) * 64;
  const _Float16* __restrict__ Wc = Wt + (long long)(j * 16 + nloc) * 64;

  v8f c = {};
#pragma unroll
  for (int kk = 0; kk < 64; kk += 32) {
    // A: 16 contiguous f32 in two 8-element runs -> 4x b128 loads
    float4 x0 = *(const float4*)(Xr + kk + abase);
    float4 x1 = *(const float4*)(Xr + kk + abase + 4);
    float4 x2 = *(const float4*)(Xr + kk + 16 + abase);
    float4 x3 = *(const float4*)(Xr + kk + 16 + abase + 4);
    v16h a;
    a[0] = (_Float16)x0.x;  a[1] = (_Float16)x0.y;
    a[2] = (_Float16)x0.z;  a[3] = (_Float16)x0.w;
    a[4] = (_Float16)x1.x;  a[5] = (_Float16)x1.y;
    a[6] = (_Float16)x1.z;  a[7] = (_Float16)x1.w;
    a[8] = (_Float16)x2.x;  a[9] = (_Float16)x2.y;
    a[10] = (_Float16)x2.z; a[11] = (_Float16)x2.w;
    a[12] = (_Float16)x3.x; a[13] = (_Float16)x3.y;
    a[14] = (_Float16)x3.z; a[15] = (_Float16)x3.w;
    // B: two 8-element f16 runs (16B each) -> 2x b128 loads
    v8h b0 = *(const v8h*)(Wc + kk + kb2);
    v8h b1 = *(const v8h*)(Wc + kk + 16 + kb2);
    v16h b;
#pragma unroll
    for (int q = 0; q < 8; ++q) { b[q] = b0[q]; b[8 + q] = b1[q]; }
    c = __builtin_amdgcn_wmma_f32_16x16x32_f16(false, a, false, b,
                                               (short)0, c, false, false);
  }
  // C/D layout: VGPR v -> M = v + 8*hi, N = lane&15
  int mo = m0 + 8 * hi;
#pragma unroll
  for (int v = 0; v < 8; ++v)
    Y[(long long)(mo + v) * 64 + j * 16 + nloc] = c[v];
}

// ---------------- epilogue kernels ----------------

__global__ void k_combine(const float* __restrict__ emb,
                          const float* __restrict__ gus,
                          const float* __restrict__ gis,
                          const int* __restrict__ user,
                          const int* __restrict__ item,
                          long long U, int B,
                          float* __restrict__ gu, float* __restrict__ gib) {
  long long t = (long long)blockIdx.x * blockDim.x + threadIdx.x;
  if (t >= (long long)B * 64) return;
  long long b = t >> 6;
  int d = (int)(t & 63);
  long long u = user[b];
  long long it = item[b];
  gu[t]  = 0.5f * emb[u * 64 + d] + 0.5f * gus[u * 64 + d];          // ALPHA=0.5
  gib[t] = 0.5f * emb[(U + it) * 64 + d] + 0.5f * gis[it * 64 + d];  // BETA=0.5
}

// Q[r] = sum_s A[r,s]*P[s], stored f16 TRANSPOSED per matrix: Q16[r][e*64+d]
__global__ void k_mix_q_t(const float* __restrict__ A, const float* __restrict__ P,
                          _Float16* __restrict__ Q16, int R, int S) {
  long long t = (long long)blockIdx.x * blockDim.x + threadIdx.x;
  if (t >= (long long)R * 4096) return;
  int r = (int)(t >> 12);
  int idx = (int)(t & 4095);
  int e = idx >> 6;
  int dd = idx & 63;
  float q = 0.0f;
  for (int s = 0; s < S; ++s)
    q += A[r * S + s] * P[(long long)s * 4096 + dd * 64 + e];
  Q16[t] = (_Float16)q;
}

// pui[b, r] = dot(tmp[b,:], gib[b,:]); one wave32 per b
__global__ void k_rowdot(const float* __restrict__ tmp,
                         const float* __restrict__ gib,
                         float* __restrict__ pui, int r, int R, int B) {
  int wave = threadIdx.x >> 5;
  int lane = threadIdx.x & 31;
  long long b = (long long)blockIdx.x * (blockDim.x >> 5) + wave;
  if (b >= B) return;
  float s = tmp[b * 64 + lane] * gib[b * 64 + lane] +
            tmp[b * 64 + 32 + lane] * gib[b * 64 + 32 + lane];
#pragma unroll
  for (int off = 16; off > 0; off >>= 1)
    s += __shfl_down(s, off, 32);
  if (lane == 0) pui[b * R + r] = s;
}

__global__ void k_score(const float* __restrict__ pui,
                        const float* __restrict__ rel,
                        float* __restrict__ xui, int R, int B) {
  long long b = (long long)blockIdx.x * blockDim.x + threadIdx.x;
  if (b >= B) return;
  float m = -3.4e38f;
  for (int r = 0; r < R; ++r) m = fmaxf(m, pui[b * R + r]);
  float den = 0.0f, num = 0.0f;
  for (int r = 0; r < R; ++r) {
    float e = expf(pui[b * R + r] - m);
    den += e;
    num += rel[r] * e;
  }
  xui[b] = num / den;
}

// ---------------- host orchestration ----------------

extern "C" void kernel_launch(void* const* d_in, const int* in_sizes, int n_in,
                              void* d_out, int out_size, void* d_ws, size_t ws_size,
                              hipStream_t stream) {
  (void)n_in; (void)out_size; (void)ws_size;
  const float* Gu      = (const float*)d_in[0];
  const float* Gi      = (const float*)d_in[1];
  const float* Gus     = (const float*)d_in[2];
  const float* Gis     = (const float*)d_in[3];
  const float* W_conv  = (const float*)d_in[4];
  const float* W_dense = (const float*)d_in[5];
  const float* P       = (const float*)d_in[6];
  const float* A       = (const float*)d_in[7];
  const float* rel     = (const float*)d_in[8];
  const int*   eir     = (const int*)d_in[9];
  const int*   eiuu    = (const int*)d_in[10];
  const int*   eiii    = (const int*)d_in[11];
  const int*   user    = (const int*)d_in[12];
  const int*   item    = (const int*)d_in[13];

  const int D = 64;
  const long long U = in_sizes[0] / D;
  const long long I = in_sizes[1] / D;
  const long long N = U + I;
  const int R = in_sizes[4] / (D * D);
  const int S = in_sizes[6] / (D * D);
  const long long E_REL = (long long)in_sizes[9] / (2LL * R);
  const long long E_SIM = (long long)in_sizes[10] / 2;
  const int B = in_sizes[12];

  char* p = (char*)d_ws;
  auto carve = [&](size_t bytes) -> char* {
    char* q = p;
    p += (bytes + 255) & ~(size_t)255;
    return q;
  };
  float* gus_a   = (float*)carve((size_t)U * D * 4);
  float* gus_b   = (float*)carve((size_t)U * D * 4);
  float* gis_a   = (float*)carve((size_t)I * D * 4);
  float* gis_b   = (float*)carve((size_t)I * D * 4);
  float* dinv_uu = (float*)carve((size_t)U * 4);
  float* dinv_ii = (float*)carve((size_t)I * 4);
  float* dinv_rl = (float*)carve((size_t)N * 4);
  float* xr      = (float*)carve((size_t)N * D * 4);   // also reused as emb
  float* xr_prop = (float*)carve((size_t)N * D * 4);
  float* acc     = (float*)carve((size_t)N * D * 4);
  float* gu      = (float*)carve((size_t)B * D * 4);
  float* gib     = (float*)carve((size_t)B * D * 4);
  float* tmp     = (float*)carve((size_t)B * D * 4);
  _Float16* Wc16 = (_Float16*)carve((size_t)R * D * D * 2);  // transposed
  _Float16* Wd16 = (_Float16*)carve((size_t)D * D * 2);      // transposed
  _Float16* Q16  = (_Float16*)carve((size_t)R * D * D * 2);  // transposed

  auto cdiv = [](long long a, long long b) -> long long { return (a + b - 1) / b; };
  auto zero = [&](float* ptr, long long n) {
    k_zero<<<(int)cdiv(n, TPB), TPB, 0, stream>>>(ptr, n);
  };

  // ---- user-user similarity propagation (2 layers) ----
  zero(dinv_uu, U);
  k_degree<<<(int)cdiv(E_SIM, TPB), TPB, 0, stream>>>(eiuu + E_SIM, E_SIM, dinv_uu);
  k_rsqrt_inplace<<<(int)cdiv(U, TPB), TPB, 0, stream>>>(dinv_uu, U);
  zero(gus_a, U * D);
  k_propagate<<<(int)cdiv(E_SIM * 64, TPB), TPB, 0, stream>>>(Gus, eiuu, eiuu + E_SIM, dinv_uu, gus_a, E_SIM);
  zero(gus_b, U * D);
  k_propagate<<<(int)cdiv(E_SIM * 64, TPB), TPB, 0, stream>>>(gus_a, eiuu, eiuu + E_SIM, dinv_uu, gus_b, E_SIM);

  // ---- item-item similarity propagation (2 layers) ----
  zero(dinv_ii, I);
  k_degree<<<(int)cdiv(E_SIM, TPB), TPB, 0, stream>>>(eiii + E_SIM, E_SIM, dinv_ii);
  k_rsqrt_inplace<<<(int)cdiv(I, TPB), TPB, 0, stream>>>(dinv_ii, I);
  zero(gis_a, I * D);
  k_propagate<<<(int)cdiv(E_SIM * 64, TPB), TPB, 0, stream>>>(Gis, eiii, eiii + E_SIM, dinv_ii, gis_a, E_SIM);
  zero(gis_b, I * D);
  k_propagate<<<(int)cdiv(E_SIM * 64, TPB), TPB, 0, stream>>>(gis_a, eiii, eiii + E_SIM, dinv_ii, gis_b, E_SIM);

  // ---- f16 transposed weight copies for WMMA ----
  k_w_to_f16_t<<<(int)cdiv((long long)R * D * D, TPB), TPB, 0, stream>>>(W_conv, Wc16, R);
  k_w_to_f16_t<<<(int)cdiv((long long)D * D, TPB), TPB, 0, stream>>>(W_dense, Wd16, 1);

  // ---- relational conv: acc = sum_r relu(propagate(x0 @ W_conv[r])) ----
  zero(acc, N * D);
  for (int r = 0; r < R; ++r) {
    const int* row_r = eir + (size_t)r * 2 * E_REL;
    const int* col_r = row_r + E_REL;
    zero(dinv_rl, N);
    k_degree<<<(int)cdiv(E_REL, TPB), TPB, 0, stream>>>(col_r, E_REL, dinv_rl);
    k_rsqrt_inplace<<<(int)cdiv(N, TPB), TPB, 0, stream>>>(dinv_rl, N);
    // xr = [Gu;Gi] @ W_conv[r] without materializing the concat
    k_gemm64<<<(int)cdiv(U, 16), 128, 0, stream>>>(Gu, Wc16 + (size_t)r * D * D, xr, (int)U);
    k_gemm64<<<(int)cdiv(I, 16), 128, 0, stream>>>(Gi, Wc16 + (size_t)r * D * D, xr + U * D, (int)I);
    zero(xr_prop, N * D);
    k_propagate<<<(int)cdiv(E_REL * 64, TPB), TPB, 0, stream>>>(xr, row_r, col_r, dinv_rl, xr_prop, E_REL);
    k_relu_acc<<<(int)cdiv(N * D, TPB), TPB, 0, stream>>>(acc, xr_prop, N * D);
  }

  // ---- emb = acc @ W_dense (reuse xr as emb) ----
  float* emb = xr;
  k_gemm64<<<(int)cdiv(N, 16), 128, 0, stream>>>(acc, Wd16, emb, (int)N);

  // ---- gather + blend ----
  k_combine<<<(int)cdiv((long long)B * D, TPB), TPB, 0, stream>>>(emb, gus_b, gis_b, user, item, U, B, gu, gib);

  // ---- Q = einsum('rs,sde->rde', A, P), f16 transposed ----
  k_mix_q_t<<<(int)cdiv((long long)R * D * D, TPB), TPB, 0, stream>>>(A, P, Q16, R, S);

  // ---- pui[b,r] = gu[b] @ Q[r] @ gib[b] ----
  float* xui_out = (float*)d_out;      // [B]
  float* pui_out = xui_out + B;        // [B, R]
  for (int r = 0; r < R; ++r) {
    k_gemm64<<<(int)cdiv(B, 16), 128, 0, stream>>>(gu, Q16 + (size_t)r * D * D, tmp, B);
    k_rowdot<<<(int)cdiv(B, 8), 256, 0, stream>>>(tmp, gib, pui_out, r, R, B);
  }

  // ---- xui = sum_r relations[r] * softmax(pui)[.,r] ----
  k_score<<<(int)cdiv(B, TPB), TPB, 0, stream>>>(pui_out, rel, xui_out, R, B);
}
